// AktMilktModel_3745211482579
// MI455X (gfx1250) — compile-verified, matrix operand order checked by reading, not run on hardware
//
#include <hip/hip_runtime.h>
#include <stdint.h>

// ---------------------------------------------------------------------------
// Problem constants (match reference)
// ---------------------------------------------------------------------------
#define Bc   32
#define Sc   512
#define Dc   256
#define Hc   8
#define DKc  32
#define DFFc 1024
#define NBc  2

// Try the CDNA5 async global->LDS path (ASYNCcnt-tracked, ISA 10.x / 15.18.3).
// Set to 0 to fall back to plain vector load + LDS store.
#ifndef USE_ASYNC_LDS
#define USE_ASYNC_LDS 1
#endif

// ---------------------------------------------------------------------------
// WMMA types / helpers (CDNA5 gfx1250, wave32)
// ---------------------------------------------------------------------------
typedef __attribute__((ext_vector_type(16))) _Float16 v16h;
typedef __attribute__((ext_vector_type(8)))  float    v8f;

union Frag {
    uint32_t u[8];
    v16h     v;
};

static __device__ inline uint16_t f2h(float f) {
    _Float16 h = (_Float16)f;
    return __builtin_bit_cast(uint16_t, h);
}
static __device__ inline uint32_t pack2(float lo, float hi) {
    return (uint32_t)f2h(lo) | ((uint32_t)f2h(hi) << 16);
}

// 16-byte global -> LDS stage. Async variant issues GLOBAL_LOAD_ASYNC_TO_LDS_B128
// (per-lane, EXEC-masked, tracked with ASYNCcnt). The LDS operand is the low 32
// bits of the generic pointer (== LDS byte offset per the aperture mapping).
static __device__ inline void stage16B(uint16_t* lds, const uint16_t* g) {
#if USE_ASYNC_LDS
    asm volatile("global_load_async_to_lds_b128 %0, %1, off"
                 :: "v"((uint32_t)(uintptr_t)lds),
                    "v"((uint64_t)(uintptr_t)g)
                 : "memory");
#else
    *(uint4*)lds = *(const uint4*)g;
#endif
}

template <int N>
static __device__ inline void wait_async() {
#if USE_ASYNC_LDS
    asm volatile("s_wait_asynccnt %0" :: "i"(N) : "memory");
#endif
}

// ---------------------------------------------------------------------------
// Elementwise fp32 -> fp16 convert (8 elems/thread, b128 in / b128 out)
// ---------------------------------------------------------------------------
__global__ __launch_bounds__(256)
void cvt_f16_kernel(const float* __restrict__ in, uint16_t* __restrict__ out)
{
    size_t i8 = ((size_t)blockIdx.x * 256 + threadIdx.x) * 8;
    float4 a = *(const float4*)(in + i8);
    float4 b = *(const float4*)(in + i8 + 4);
    uint4 o;
    o.x = pack2(a.x, a.y);
    o.y = pack2(a.z, a.w);
    o.z = pack2(b.x, b.y);
    o.w = pack2(b.z, b.w);
    *(uint4*)(out + i8) = o;
}

// ---------------------------------------------------------------------------
// Weight transpose + convert: W[K,N] fp32 -> Wt[N,K] fp16 (K-pairs packed)
// ---------------------------------------------------------------------------
__global__ __launch_bounds__(256)
void cvt_wt_kernel(const float* __restrict__ W, uint16_t* __restrict__ Wt,
                   int K, int N)
{
    size_t t = (size_t)blockIdx.x * 256 + threadIdx.x;   // one k-pair per thread
    int kh = K >> 1;
    int n  = (int)(t / kh);
    int k  = ((int)(t % kh)) * 2;
    float w0 = W[(size_t)k * N + n];
    float w1 = W[(size_t)(k + 1) * N + n];
    *(uint32_t*)&Wt[(size_t)n * K + k] = pack2(w0, w1);
}

// ---------------------------------------------------------------------------
// GEMM: C[M,N] = op(A16[M,K] @ Wt16[N,K]^T + bias[N]).  f16 WMMA, f32 acc.
// 256 threads = 8 waves; block tile 128x128; wave tile 32x64 (2x4 WMMA quad).
// Double-buffered LDS, async global->LDS staging.
// ---------------------------------------------------------------------------
#define GM 128
#define GN 128
#define LDP 40   // LDS row pitch (halves): 80B rows -> 16B-aligned segs, no bank conflicts

__global__ __launch_bounds__(256)
void gemm16_kernel(const uint16_t* __restrict__ A, const uint16_t* __restrict__ Wt,
                   const float* __restrict__ bias, float* __restrict__ C,
                   uint16_t* __restrict__ C16, int M, int N, int K, int relu)
{
    __shared__ __align__(16) uint16_t As[2][GM][LDP];   // [buf][m][k]
    __shared__ __align__(16) uint16_t Ws[2][GN][LDP];   // [buf][n][k]

    const int tid  = threadIdx.x;
    const int lane = tid & 31;
    const int wave = tid >> 5;
    const int wm   = (wave & 3) * 32;        // wave row offset (4 waves x 32 = 128)
    const int wn   = (wave >> 2) * 64;       // wave col offset (2 waves x 64 = 128)
    const int m0   = blockIdx.x * GM;
    const int n0   = blockIdx.y * GN;
    const int rlo  = lane & 15;
    const int kb   = (lane >> 4) * 8;

    v8f acc[2][4] = {};

    // stage one 128x32 A tile + 128x32 Wt tile into buffer b (4 async/thread)
    auto stage = [&](int ti, int b) {
        const int k0 = ti * 32;
        #pragma unroll
        for (int i = 0; i < 2; ++i) {
            int s = tid + i * 256;           // 512 slots: row(7b) x seg(2b)
            int row = s >> 2, seg = (s & 3) * 8;
            stage16B(&As[b][row][seg], A  + (size_t)(m0 + row) * K + k0 + seg);
            stage16B(&Ws[b][row][seg], Wt + (size_t)(n0 + row) * K + k0 + seg);
        }
    };

    auto compute = [&](int b) {
        Frag af[2], bf[4];
        #pragma unroll
        for (int v8i = 0; v8i < 8; ++v8i) {
            int kk = (v8i >> 2) * 16 + kb + 2 * (v8i & 3);
            #pragma unroll
            for (int mt = 0; mt < 2; ++mt)
                af[mt].u[v8i] = *(const uint32_t*)&As[b][wm + mt * 16 + rlo][kk];
            #pragma unroll
            for (int nt = 0; nt < 4; ++nt)
                bf[nt].u[v8i] = *(const uint32_t*)&Ws[b][wn + nt * 16 + rlo][kk];
        }
        #pragma unroll
        for (int mt = 0; mt < 2; ++mt)
            #pragma unroll
            for (int nt = 0; nt < 4; ++nt)
                acc[mt][nt] = __builtin_amdgcn_wmma_f32_16x16x32_f16(
                    false, af[mt].v, false, bf[nt].v, (short)0, acc[mt][nt], false, false);
    };

    const int nk = K / 32;
    stage(0, 0);
    for (int ti = 0; ti < nk - 1; ++ti) {
        stage(ti + 1, (ti + 1) & 1);   // prefetch next tile while current in flight
        wait_async<4>();               // tile ti complete (async ops retire in order)
        __syncthreads();
        compute(ti & 1);
        __syncthreads();               // all waves done reading before restage
    }
    wait_async<0>();
    __syncthreads();
    compute((nk - 1) & 1);

    // C layout: VGPR r, lanes 0-15 -> M=r, lanes 16-31 -> M=8+r, N=lane&15
    const int mtop = (lane >> 4) * 8;
    #pragma unroll
    for (int mt = 0; mt < 2; ++mt) {
        #pragma unroll
        for (int nt = 0; nt < 4; ++nt) {
            int gcol = n0 + wn + nt * 16 + rlo;
            float bv = bias[gcol];
            #pragma unroll
            for (int r = 0; r < 8; ++r) {
                int grow = m0 + wm + mt * 16 + mtop + r;
                float val = acc[mt][nt][r] + bv;
                if (relu) val = fmaxf(val, 0.f);
                C[(size_t)grow * N + gcol] = val;
                if (C16) C16[(size_t)grow * N + gcol] = f2h(val);
            }
        }
    }
}

// ---------------------------------------------------------------------------
// Attention with distance-decay (AKT "monotonic attention").
// One wave32 per (b, h, 16-query-row tile). qh16/vh16 are f16 [B,S,D] with
// head h at columns [h*32, h*32+32).
// ---------------------------------------------------------------------------
__global__ __launch_bounds__(32)
void attn_kernel(const uint16_t* __restrict__ qh16, const uint16_t* __restrict__ kh16,
                 const uint16_t* __restrict__ vh16, const float* __restrict__ gam,
                 float* __restrict__ o, int mask_flag)
{
    __shared__ float sc[16 * Sc];   // 32 KB: scores, then attn probabilities

    const int qt = blockIdx.x, h = blockIdx.y, b = blockIdx.z;
    const int lane = threadIdx.x;
    const int rlo = lane & 15;
    const int kb  = (lane >> 4) * 8;
    const float scale = 0.17677669529663687f;   // 1/sqrt(DK)

    const uint16_t* qbase = qh16 + (size_t)(b * Sc) * Dc + h * DKc;
    const uint16_t* kbase = kh16 + (size_t)(b * Sc) * Dc + h * DKc;
    const uint16_t* vbase = vh16 + (size_t)(b * Sc) * Dc + h * DKc;

    // Q fragment: 16 rows x 32 (DK) — dword pair loads, K contiguous
    Frag qf;
    {
        const uint16_t* qrow = qbase + (size_t)(qt * 16 + rlo) * Dc;
        #pragma unroll
        for (int v8i = 0; v8i < 8; ++v8i) {
            int k0 = (v8i >> 2) * 16 + kb + 2 * (v8i & 3);
            qf.u[v8i] = *(const uint32_t*)&qrow[k0];
        }
    }

    // --- score strip: S[m][j*16+n] = (Q K^T)/sqrt(DK) ---
    for (int j = 0; j < Sc / 16; ++j) {
        Frag ktf;   // B-matrix 32x16: B[k][n] = K[j*16+n][k]
        const uint16_t* krow = kbase + (size_t)(j * 16 + rlo) * Dc;
        #pragma unroll
        for (int v8i = 0; v8i < 8; ++v8i) {
            int k0 = (v8i >> 2) * 16 + kb + 2 * (v8i & 3);
            ktf.u[v8i] = *(const uint32_t*)&krow[k0];
        }
        v8f s = {};
        s = __builtin_amdgcn_wmma_f32_16x16x32_f16(false, qf.v, false, ktf.v,
                                                   (short)0, s, false, false);
        const int mtop = (lane >> 4) * 8;
        #pragma unroll
        for (int r = 0; r < 8; ++r)
            sc[(mtop + r) * Sc + j * 16 + rlo] = s[r] * scale;
    }
    __syncthreads();

    // --- per-row serial pass (lanes 0..15, one query row each) ---
    if (lane < 16) {
        const int m  = lane;
        const int qi = qt * 16 + m;
        const int limit = mask_flag ? qi : (qi - 1);   // last attended col (incl.)
        float* row = sc + m * Sc;

        float mx = -3.0e38f;
        for (int j = 0; j <= limit; ++j) mx = fmaxf(mx, row[j]);
        float sum1 = 0.f;
        for (int j = 0; j <= limit; ++j) sum1 += __expf(row[j] - mx);
        const float inv1 = (limit >= 0 && sum1 > 0.f) ? 1.f / sum1 : 0.f;
        const float tot  = (limit >= 0) ? 1.f : 0.f;   // sum(p) over masked region

        float g  = gam[h];
        float sp = (g > 20.f) ? g : log1pf(__expf(g));  // softplus
        const float gamma = -sp;

        float cum = 0.f, mx2 = -3.0e38f;
        for (int j = 0; j < Sc; ++j) {
            float sv = row[j], s2;
            if (j <= limit) {
                float p = __expf(sv - mx) * inv1;
                cum += p;
                float pe = fabsf((float)(qi - j));
                float dist = sqrtf(fmaxf((tot - cum) * pe, 0.f));
                float te = __expf(dist * gamma);
                te = fminf(fmaxf(te, 1e-5f), 1e5f);
                s2 = sv * te;
            } else {
                s2 = -1e32f;
            }
            row[j] = s2;
            mx2 = fmaxf(mx2, s2);
        }
        float sum2 = 0.f;
        for (int j = 0; j < Sc; ++j) sum2 += __expf(row[j] - mx2);
        const float inv2 = 1.f / sum2;
        const bool zp = (mask_flag == 0) && (qi == 0);   // zero_pad of first query row
        for (int j = 0; j < Sc; ++j) {
            float a = __expf(row[j] - mx2) * inv2;
            row[j] = zp ? 0.f : a;
        }
    }
    __syncthreads();

    // --- O = attn(16x512) @ V(512x32), 16 K-chunks of 32 ---
    v8f acc0 = {}, acc1 = {};
    for (int kc = 0; kc < Sc / 32; ++kc) {
        Frag af;
        const float* arow = sc + rlo * Sc + kc * 32;
        #pragma unroll
        for (int v8i = 0; v8i < 8; ++v8i) {
            int k0 = (v8i >> 2) * 16 + kb + 2 * (v8i & 3);
            af.u[v8i] = pack2(arow[k0], arow[k0 + 1]);
        }
        Frag vf0, vf1;   // B[k][n] = V[key=kc*32+k][dk = n (+16)]
        #pragma unroll
        for (int v8i = 0; v8i < 8; ++v8i) {
            int k0 = (v8i >> 2) * 16 + kb + 2 * (v8i & 3);
            const uint16_t* r0 = vbase + (size_t)(kc * 32 + k0) * Dc;
            const uint16_t* r1 = r0 + Dc;
            vf0.u[v8i] = (uint32_t)r0[rlo]      | ((uint32_t)r1[rlo]      << 16);
            vf1.u[v8i] = (uint32_t)r0[16 + rlo] | ((uint32_t)r1[16 + rlo] << 16);
        }
        acc0 = __builtin_amdgcn_wmma_f32_16x16x32_f16(false, af.v, false, vf0.v,
                                                      (short)0, acc0, false, false);
        acc1 = __builtin_amdgcn_wmma_f32_16x16x32_f16(false, af.v, false, vf1.v,
                                                      (short)0, acc1, false, false);
    }
    {
        float* obase = o + (size_t)(b * Sc) * Dc + h * DKc;
        const int mtop = (lane >> 4) * 8;
        #pragma unroll
        for (int r = 0; r < 8; ++r) {
            float* orow = obase + (size_t)(qt * 16 + mtop + r) * Dc;
            orow[rlo]      = acc0[r];
            orow[16 + rlo] = acc1[r];
        }
    }
}

// ---------------------------------------------------------------------------
// Fused residual + LayerNorm: out = LN(x + r) * g + b.  One wave32 per row.
// ---------------------------------------------------------------------------
__global__ __launch_bounds__(256)
void add_ln_kernel(const float* __restrict__ x, const float* __restrict__ r,
                   const float* __restrict__ g, const float* __restrict__ bta,
                   float* __restrict__ out)
{
    const int lane = threadIdx.x & 31;
    const int wv   = threadIdx.x >> 5;
    const size_t row = (size_t)blockIdx.x * 8 + wv;
    const float* xr = x + row * Dc;
    const float* rr = r + row * Dc;

    float v[8];
    float s = 0.f;
    #pragma unroll
    for (int i = 0; i < 8; ++i) { v[i] = xr[lane + i * 32] + rr[lane + i * 32]; s += v[i]; }
    #pragma unroll
    for (int off = 16; off > 0; off >>= 1) s += __shfl_xor(s, off, 32);
    const float mu = s * (1.f / Dc);

    float var = 0.f;
    #pragma unroll
    for (int i = 0; i < 8; ++i) { float d = v[i] - mu; var += d * d; }
    #pragma unroll
    for (int off = 16; off > 0; off >>= 1) var += __shfl_xor(var, off, 32);
    const float rs = rsqrtf(var * (1.f / Dc) + 1e-5f);

    float* orow = out + row * Dc;
    #pragma unroll
    for (int i = 0; i < 8; ++i) {
        int c = lane + i * 32;
        orow[c] = (v[i] - mu) * rs * g[c] + bta[c];
    }
}

// ---------------------------------------------------------------------------
// Host orchestration
// ---------------------------------------------------------------------------
struct LayerP {
    const float *Wk, *bk, *Wv, *bv, *Wo, *bo, *gammas;
    const float *ln1g, *ln1b, *W1, *b1, *W2, *b2, *ln2g, *ln2b;
};

static LayerP slice_params(void* const* P, int i) {
    LayerP p;
    p.Wk   = (const float*)P[0]  + (size_t)i * Dc * Dc;
    p.bk   = (const float*)P[1]  + (size_t)i * Dc;
    p.Wv   = (const float*)P[2]  + (size_t)i * Dc * Dc;
    p.bv   = (const float*)P[3]  + (size_t)i * Dc;
    p.Wo   = (const float*)P[4]  + (size_t)i * Dc * Dc;
    p.bo   = (const float*)P[5]  + (size_t)i * Dc;
    p.gammas = (const float*)P[6] + (size_t)i * Hc;
    p.ln1g = (const float*)P[7]  + (size_t)i * Dc;
    p.ln1b = (const float*)P[8]  + (size_t)i * Dc;
    p.W1   = (const float*)P[9]  + (size_t)i * Dc * DFFc;
    p.b1   = (const float*)P[10] + (size_t)i * DFFc;
    p.W2   = (const float*)P[11] + (size_t)i * DFFc * Dc;
    p.b2   = (const float*)P[12] + (size_t)i * Dc;
    p.ln2g = (const float*)P[13] + (size_t)i * Dc;
    p.ln2b = (const float*)P[14] + (size_t)i * Dc;
    return p;
}

// convert A (+ transpose/convert W) then run the f16 WMMA GEMM
static void gemm_full(const float* A32, const float* W, const float* bias,
                      float* C32, uint16_t* C16, int M, int N, int K, int relu,
                      uint16_t* act16, uint16_t* wt16, hipStream_t s)
{
    cvt_f16_kernel<<<((size_t)M * K) / 2048, 256, 0, s>>>(A32, act16);
    cvt_wt_kernel<<<((size_t)N * K) / 512, 256, 0, s>>>(W, wt16, K, N);
    dim3 g(M / GM, N / GN);
    gemm16_kernel<<<g, 256, 0, s>>>(act16, wt16, bias, C32, C16, M, N, K, relu);
}

struct Bufs {
    float *y, *qh32, *vh32, *o, *t1, *hid;
    uint16_t *act16, *wt16, *qh16, *vh16;
};

// query==key in every reference call (kq_same=True), so kh aliases qh.
static void run_block(const LayerP& p, int mask_flag, float* q, float* v, int apply_pos,
                      Bufs& w, hipStream_t s)
{
    const int M = Bc * Sc;

    gemm_full(q, p.Wk, p.bk, w.qh32, w.qh16, M, Dc, Dc, 0, w.act16, w.wt16, s);
    gemm_full(v, p.Wv, p.bv, w.vh32, w.vh16, M, Dc, Dc, 0, w.act16, w.wt16, s);

    dim3 gA(Sc / 16, Hc, Bc);
    attn_kernel<<<gA, 32, 0, s>>>(w.qh16, w.qh16, w.vh16, p.gammas, w.o, mask_flag);

    float* proj = w.qh32;   // qh32 dead after attention
    gemm_full(w.o, p.Wo, p.bo, proj, nullptr, M, Dc, Dc, 0, w.act16, w.wt16, s);

    float* ln1out = apply_pos ? w.t1 : q;
    add_ln_kernel<<<M / 8, 256, 0, s>>>(q, proj, p.ln1g, p.ln1b, ln1out);

    if (apply_pos) {
        gemm_full(w.t1, p.W1, p.b1, w.hid, nullptr, M, DFFc, Dc, 1, w.act16, w.wt16, s);
        float* f2 = w.vh32;  // vh32 dead after attention
        gemm_full(w.hid, p.W2, p.b2, f2, nullptr, M, Dc, DFFc, 0, w.act16, w.wt16, s);
        add_ln_kernel<<<M / 8, 256, 0, s>>>(w.t1, f2, p.ln2g, p.ln2b, q);
    }
}

extern "C" void kernel_launch(void* const* d_in, const int* in_sizes, int n_in,
                              void* d_out, int out_size, void* d_ws, size_t ws_size,
                              hipStream_t stream)
{
    (void)in_sizes; (void)n_in; (void)out_size; (void)ws_size;

    const float* q_embed  = (const float*)d_in[0];
    const float* qa_embed = (const float*)d_in[1];
    void* const* P1 = d_in + 2;    // 15 arrays, leading dim NB
    void* const* P2 = d_in + 17;   // 15 arrays, leading dim 2*NB

    const size_t BSD = (size_t)Bc * Sc * Dc;          // 4,194,304
    const size_t BSF = (size_t)Bc * Sc * DFFc;        // 16,777,216

    float* x = (float*)d_out;      // x lives in d_out the whole time

    char* base = (char*)d_ws;
    Bufs w;
    w.y    = (float*)base;  base += BSD * 4;
    w.qh32 = (float*)base;  base += BSD * 4;
    w.vh32 = (float*)base;  base += BSD * 4;
    w.o    = (float*)base;  base += BSD * 4;
    w.t1   = (float*)base;  base += BSD * 4;
    w.hid  = (float*)base;  base += BSF * 4;
    w.act16 = (uint16_t*)base; base += BSF * 2;        // max GEMM input (M x DFF)
    w.wt16  = (uint16_t*)base; base += (size_t)DFFc * Dc * 2;
    w.qh16  = (uint16_t*)base; base += BSD * 2;
    w.vh16  = (uint16_t*)base; base += BSD * 2;

    hipMemcpyAsync(x,   q_embed,  BSD * sizeof(float), hipMemcpyDeviceToDevice, stream);
    hipMemcpyAsync(w.y, qa_embed, BSD * sizeof(float), hipMemcpyDeviceToDevice, stream);

    for (int i = 0; i < NBc; ++i) {
        LayerP p = slice_params(P1, i);
        run_block(p, /*mask_flag=*/1, w.y, w.y, /*apply_pos=*/1, w, stream);
    }
    for (int i = 0; i < 2 * NBc; ++i) {
        LayerP p = slice_params(P2, i);
        if ((i & 1) == 0)
            run_block(p, 1, x, x, /*apply_pos=*/0, w, stream);
        else
            run_block(p, 0, x, w.y, /*apply_pos=*/1, w, stream);
    }
}
